// LSTMnet_25005299597861
// MI455X (gfx1250) — matrix operand order repaired and checked by reading, not compiled, not run on hardware
//
#include <hip/hip_runtime.h>
#include <hip/hip_bf16.h>

// ---------------------------------------------------------------------------
// LSTM net on MI455X (gfx1250): bf16 WMMA for all GEMMs, fp32 accumulate,
// fp32 elementwise cell math. Phases:
//   1) prep: cast Wx/Wh/fc_w to bf16, bias = bW+bU, zero h0/c0
//   2) xg GEMM: xg[s,b,g] = sum_i x[b,s,i]*Wx[g,i] + bias[g]   (WMMA)
//   3) 1024x step kernel: gates = xg[t] + h.Wh^T ; cell update
//      (WMMA fused, async-to-LDS double-buffered pipeline)
//   4) head: out = hT.fc_w^T + fc_b                             (WMMA)
// ---------------------------------------------------------------------------

typedef __attribute__((ext_vector_type(16))) __bf16 v16bf;
typedef __attribute__((ext_vector_type(8)))  __bf16 v8bf;
typedef __attribute__((ext_vector_type(8)))  float  v8f;
typedef __attribute__((ext_vector_type(4)))  float  v4f;

constexpr int kB   = 128;    // batch
constexpr int kS   = 1024;   // sequence
constexpr int kI   = 512;    // input features
constexpr int kH   = 1024;   // hidden
constexpr int kO   = 128;    // output features
constexpr int kG4H = 4 * kH; // 4096 gate width

constexpr int KC  = 64;      // staged k-chunk (2 WMMA k-steps)
constexpr int NKC = kH / KC; // 16 chunks per step

__device__ __forceinline__ float sigmoidf_(float x) {
    return 1.0f / (1.0f + __expf(-x));
}
__device__ __forceinline__ float tanhf_(float x) {
    float e = __expf(2.0f * x);
    return (e - 1.0f) / (e + 1.0f);
}

// Async copy engine: 16B global -> LDS per lane, tracked by ASYNCcnt.
__device__ __forceinline__ void async_copy16(uint32_t lds_addr,
                                             const void* gaddr) {
    asm volatile("global_load_async_to_lds_b128 %0, %1, off"
                 :: "v"(lds_addr), "v"((uint64_t)(uintptr_t)gaddr)
                 : "memory");
}
__device__ __forceinline__ void wait_async_le8() {
    asm volatile("s_wait_asynccnt 8" ::: "memory");
}
__device__ __forceinline__ void wait_async_0() {
    asm volatile("s_wait_asynccnt 0" ::: "memory");
}

// ---- WMMA fragment loaders (wave32, 16x16x32 bf16) ------------------------
// A-matrix 16x32 bf16 (ISA 7.12.2): lane L holds row M = L&15.
//   lanes 0-15 : K = k0+0..7  and k0+16..23
//   lanes 16-31: K = k0+8..15 and k0+24..31
__device__ __forceinline__ v16bf load_a_bf16(const __bf16* p, int row, int k0,
                                             int ld, int lane) {
    const __bf16* r = p + (size_t)row * ld + k0 + ((lane < 16) ? 0 : 8);
    v8bf lo = *(const v8bf*)(r);
    v8bf hi = *(const v8bf*)(r + 16);
    return __builtin_shufflevector(lo, hi, 0, 1, 2, 3, 4, 5, 6, 7,
                                   8, 9, 10, 11, 12, 13, 14, 15);
}

// Same A layout, sourced from fp32 with on-the-fly bf16 conversion.
__device__ __forceinline__ v16bf load_a_f32(const float* p, int row, int k0,
                                            int ld, int lane) {
    const float* r = p + (size_t)row * ld + k0 + ((lane < 16) ? 0 : 8);
    v4f f0 = *(const v4f*)(r);
    v4f f1 = *(const v4f*)(r + 4);
    v4f f2 = *(const v4f*)(r + 16);
    v4f f3 = *(const v4f*)(r + 20);
    v16bf a;
#pragma unroll
    for (int i = 0; i < 4; ++i) {
        a[i]      = (__bf16)f0[i];
        a[4 + i]  = (__bf16)f1[i];
        a[8 + i]  = (__bf16)f2[i];
        a[12 + i] = (__bf16)f3[i];
    }
    return a;
}

// B-matrix 32x16 bf16: lane L holds column N = L&15; lanes 0-15 hold
// K = k0+0..7 and k0+16..23, lanes 16-31 hold K = k0+8..15 and k0+24..31.
// B[k][n] = W[n][k].
__device__ __forceinline__ v16bf load_b_bf16(const __bf16* p, int colbase,
                                             int k0, int ld, int lane) {
    const __bf16* r =
        p + (size_t)(colbase + (lane & 15)) * ld + k0 + ((lane < 16) ? 0 : 8);
    v8bf lo = *(const v8bf*)(r);
    v8bf hi = *(const v8bf*)(r + 16);
    return __builtin_shufflevector(lo, hi, 0, 1, 2, 3, 4, 5, 6, 7,
                                   8, 9, 10, 11, 12, 13, 14, 15);
}

__device__ __forceinline__ v8f wmma_bf16(v16bf a, v16bf b, v8f c) {
    return __builtin_amdgcn_wmma_f32_16x16x32_bf16(
        /*neg_a=*/false, a, /*neg_b=*/false, b,
        /*c_mod=*/(short)0, c, /*reuse_a=*/false, /*reuse_b=*/false);
}

// ---------------------------------------------------------------------------
// 1) prep: weight conversion + bias fuse + state zero-init (runs every launch)
// ---------------------------------------------------------------------------
__global__ void prep_kernel(const float* __restrict__ Wx,
                            const float* __restrict__ Wh,
                            const float* __restrict__ fcw,
                            const float* __restrict__ bW,
                            const float* __restrict__ bU,
                            __bf16* Wx_b, __bf16* Wh_b, __bf16* fcw_b,
                            float* bias, __bf16* h0, float* c0) {
    int idx = blockIdx.x * blockDim.x + threadIdx.x;
    int stride = gridDim.x * blockDim.x;
    for (int i = idx; i < kG4H * kI; i += stride) Wx_b[i] = (__bf16)Wx[i];
    for (int i = idx; i < kG4H * kH; i += stride) Wh_b[i] = (__bf16)Wh[i];
    for (int i = idx; i < kO * kH;   i += stride) fcw_b[i] = (__bf16)fcw[i];
    for (int i = idx; i < kG4H;      i += stride) bias[i] = bW[i] + bU[i];
    for (int i = idx; i < kB * kH;   i += stride) {
        h0[i] = (__bf16)0.0f;
        c0[i] = 0.0f;
    }
}

// ---------------------------------------------------------------------------
// 2) xg GEMM: M = B*S = 131072 (row = b*S+s), N = 4H, K = I.
//    Block = 256 threads (8 waves); wave tile 16x64; WG tile 64x128.
// ---------------------------------------------------------------------------
__global__ void __launch_bounds__(256)
xg_gemm_kernel(const float* __restrict__ x, const __bf16* __restrict__ Wxb,
               const float* __restrict__ bias, float* __restrict__ xg) {
    const int lane = threadIdx.x & 31;
    const int wave = threadIdx.x >> 5;
    const int m0 = blockIdx.x * 64 + (wave & 3) * 16;
    const int n0 = blockIdx.y * 128 + (wave >> 2) * 64;

    v8f acc[4] = {};
    for (int k0 = 0; k0 < kI; k0 += 32) {
        v16bf a = load_a_f32(x, m0 + (lane & 15), k0, kI, lane);
#pragma unroll
        for (int j = 0; j < 4; ++j) {
            v16bf b = load_b_bf16(Wxb, n0 + j * 16, k0, kI, lane);
            acc[j] = wmma_bf16(a, b, acc[j]);
        }
    }
#pragma unroll
    for (int j = 0; j < 4; ++j) {
        int col = n0 + j * 16 + (lane & 15);
        float bv = bias[col];
#pragma unroll
        for (int r = 0; r < 8; ++r) {
            int row = m0 + r + ((lane < 16) ? 0 : 8); // row = b*S + s
            int bb = row >> 10;                       // /kS
            int ss = row & (kS - 1);
            xg[((size_t)ss * kB + bb) * kG4H + col] = acc[j][r] + bv;
        }
    }
}

// ---------------------------------------------------------------------------
// 3) fused LSTM step with async-to-LDS double buffering.
//    Block tile: all 128 batch rows x 32 hidden cols x 4 gates; 8 waves.
//    wave: m0 = (w&3)*32 rows, 16-col half = (w>>2). Per k-chunk (KC=64)
//    stage h-tile (16 KB) + 4 Wh gate strips (16 KB) via the async copy
//    engine while WMMAs consume the previous chunk from LDS.
// ---------------------------------------------------------------------------
__global__ void __launch_bounds__(256)
lstm_step_kernel(const __bf16* __restrict__ hin, const __bf16* __restrict__ Whb,
                 const float* __restrict__ xg_t, float* __restrict__ c,
                 __bf16* __restrict__ hout) {
    __shared__ __bf16 sh_a[2][kB][KC];     // 2*128*64*2B = 32 KB
    __shared__ __bf16 sh_b[2][4][32][KC];  // 2*4*32*64*2B = 32 KB

    const int tid  = threadIdx.x;
    const int lane = tid & 31;
    const int wave = tid >> 5;
    const int jblk = blockIdx.x * 32;      // hidden-column block tile
    const int m0   = (wave & 3) * 32;
    const int jh   = (wave >> 2) * 16;

    auto issue_stage = [&](int kc, int buf) {
        const int k0 = kc * KC;
        // A tile: 128 rows x 64 bf16 -> 1024 x 16B chunks (4 per thread)
#pragma unroll
        for (int i = 0; i < 4; ++i) {
            int cidx = tid + i * 256;
            int row  = cidx >> 3;
            int ck   = (cidx & 7) * 8;
            async_copy16((uint32_t)(uintptr_t)&sh_a[buf][row][ck],
                         hin + (size_t)row * kH + k0 + ck);
        }
        // B tile: 4 gates x 32 rows x 64 bf16 -> 1024 chunks (4 per thread)
#pragma unroll
        for (int i = 0; i < 4; ++i) {
            int cidx = tid + i * 256;
            int g    = cidx >> 8;
            int rr   = (cidx >> 3) & 31;
            int ck   = (cidx & 7) * 8;
            async_copy16((uint32_t)(uintptr_t)&sh_b[buf][g][rr][ck],
                         Whb + (size_t)(g * kH + jblk + rr) * kH + k0 + ck);
        }
    };

    v8f acc[4][2] = {};
    issue_stage(0, 0);

    for (int kc = 0; kc < NKC; ++kc) {
        const int buf = kc & 1;
        if (kc + 1 < NKC) {
            issue_stage(kc + 1, buf ^ 1);
            wait_async_le8();   // own stage-kc copies done; kc+1 in flight
        } else {
            wait_async_0();
        }
        __syncthreads();        // all waves' stage-kc data resident in LDS

#pragma unroll
        for (int ks = 0; ks < KC; ks += 32) {
            const int ka = ks + ((lane < 16) ? 0 : 8);
            v16bf a0, a1;
            {
                const __bf16* r = &sh_a[buf][m0 + (lane & 15)][ka];
                v8bf lo = *(const v8bf*)(r);
                v8bf hi = *(const v8bf*)(r + 16);
                a0 = __builtin_shufflevector(lo, hi, 0, 1, 2, 3, 4, 5, 6, 7,
                                             8, 9, 10, 11, 12, 13, 14, 15);
            }
            {
                const __bf16* r = &sh_a[buf][m0 + 16 + (lane & 15)][ka];
                v8bf lo = *(const v8bf*)(r);
                v8bf hi = *(const v8bf*)(r + 16);
                a1 = __builtin_shufflevector(lo, hi, 0, 1, 2, 3, 4, 5, 6, 7,
                                             8, 9, 10, 11, 12, 13, 14, 15);
            }
            const int kb = ks + ((lane < 16) ? 0 : 8);
#pragma unroll
            for (int g = 0; g < 4; ++g) {
                const __bf16* r = &sh_b[buf][g][jh + (lane & 15)][kb];
                v8bf lo = *(const v8bf*)(r);
                v8bf hi = *(const v8bf*)(r + 16);
                v16bf b = __builtin_shufflevector(lo, hi, 0, 1, 2, 3, 4, 5, 6,
                                                  7, 8, 9, 10, 11, 12, 13, 14,
                                                  15);
                acc[g][0] = wmma_bf16(a0, b, acc[g][0]);
                acc[g][1] = wmma_bf16(a1, b, acc[g][1]);
            }
        }
        __syncthreads();        // reads done before buffer is overwritten
    }

    const int j = jblk + jh + (lane & 15);
#pragma unroll
    for (int mi = 0; mi < 2; ++mi) {
#pragma unroll
        for (int r = 0; r < 8; ++r) {
            int b = m0 + mi * 16 + r + ((lane < 16) ? 0 : 8);
            const float* xr = xg_t + (size_t)b * kG4H;
            float fv = acc[0][mi][r] + xr[0 * kH + j];
            float iv = acc[1][mi][r] + xr[1 * kH + j];
            float gv = acc[2][mi][r] + xr[2 * kH + j];
            float ov = acc[3][mi][r] + xr[3 * kH + j];
            size_t ci = (size_t)b * kH + j;
            float cv = sigmoidf_(fv) * c[ci] + sigmoidf_(iv) * tanhf_(gv);
            c[ci] = cv;
            hout[ci] = (__bf16)(sigmoidf_(ov) * tanhf_(cv));
        }
    }
}

// ---------------------------------------------------------------------------
// 4) head: out[b,o] = hT . fc_w^T + fc_b ; M=128, N=128, K=1024.
// ---------------------------------------------------------------------------
__global__ void __launch_bounds__(256)
head_kernel(const __bf16* __restrict__ h, const __bf16* __restrict__ fcwb,
            const float* __restrict__ fcb, float* __restrict__ out) {
    const int lane = threadIdx.x & 31;
    const int wave = threadIdx.x >> 5;
    const int m0 = blockIdx.x * 64 + (wave & 3) * 16;
    const int n0 = (wave >> 2) * 64;

    v8f acc[4] = {};
    for (int k0 = 0; k0 < kH; k0 += 32) {
        v16bf a = load_a_bf16(h, m0 + (lane & 15), k0, kH, lane);
#pragma unroll
        for (int j = 0; j < 4; ++j) {
            v16bf b = load_b_bf16(fcwb, n0 + j * 16, k0, kH, lane);
            acc[j] = wmma_bf16(a, b, acc[j]);
        }
    }
#pragma unroll
    for (int j = 0; j < 4; ++j) {
        int col = n0 + j * 16 + (lane & 15);
        float bv = fcb[col];
#pragma unroll
        for (int r = 0; r < 8; ++r) {
            int row = m0 + r + ((lane < 16) ? 0 : 8);
            out[row * kO + col] = acc[j][r] + bv;
        }
    }
}

// ---------------------------------------------------------------------------
extern "C" void kernel_launch(void* const* d_in, const int* in_sizes, int n_in,
                              void* d_out, int out_size, void* d_ws,
                              size_t ws_size, hipStream_t stream) {
    const float* x   = (const float*)d_in[0];
    const float* Wx  = (const float*)d_in[1];
    const float* bW  = (const float*)d_in[2];
    const float* Wh  = (const float*)d_in[3];
    const float* bU  = (const float*)d_in[4];
    const float* fcw = (const float*)d_in[5];
    const float* fcb = (const float*)d_in[6];
    (void)in_sizes; (void)n_in; (void)out_size; (void)ws_size;

    char* wsp = (char*)d_ws;
    auto carve = [&](size_t bytes) -> char* {
        char* p = wsp;
        wsp += (bytes + 255) & ~(size_t)255;
        return p;
    };
    __bf16* Wx_b  = (__bf16*)carve((size_t)kG4H * kI * sizeof(__bf16));
    __bf16* Wh_b  = (__bf16*)carve((size_t)kG4H * kH * sizeof(__bf16));
    __bf16* fcw_b = (__bf16*)carve((size_t)kO * kH * sizeof(__bf16));
    float*  bias  = (float*)carve((size_t)kG4H * sizeof(float));
    __bf16* hbuf0 = (__bf16*)carve((size_t)kB * kH * sizeof(__bf16));
    __bf16* hbuf1 = (__bf16*)carve((size_t)kB * kH * sizeof(__bf16));
    float*  cbuf  = (float*)carve((size_t)kB * kH * sizeof(float));
    float*  xg    = (float*)carve((size_t)kS * kB * kG4H * sizeof(float));

    // 1) convert + init
    prep_kernel<<<1024, 256, 0, stream>>>(Wx, Wh, fcw, bW, bU, Wx_b, Wh_b,
                                          fcw_b, bias, hbuf0, cbuf);

    // 2) input-projection GEMM: grid (M/64, N/128)
    xg_gemm_kernel<<<dim3((kB * kS) / 64, kG4H / 128), 256, 0, stream>>>(
        x, Wx_b, bias, xg);

    // 3) sequential scan: one launch per timestep (graph-captured)
    for (int t = 0; t < kS; ++t) {
        const __bf16* hin = (t & 1) ? hbuf1 : hbuf0;
        __bf16* hout      = (t & 1) ? hbuf0 : hbuf1;
        lstm_step_kernel<<<kH / 32, 256, 0, stream>>>(
            hin, Wh_b, xg + (size_t)t * kB * kG4H, cbuf, hout);
    }

    // 4) head: S even -> final h is in hbuf0
    head_kernel<<<dim3(kB / 64, 1), 256, 0, stream>>>(hbuf0, fcw_b, fcb,
                                                      (float*)d_out);
}
